// EarthSpecificBlockBase_16595753631907
// MI455X (gfx1250) — matrix operand order repaired and checked by reading, not compiled, not run on hardware
//
#include <hip/hip_runtime.h>
#include <hip/hip_bf16.h>

typedef __attribute__((ext_vector_type(16))) __bf16 v16bf;
typedef __attribute__((ext_vector_type(8)))  __bf16 v8bf;
typedef __attribute__((ext_vector_type(8)))  float  v8f;

// ---- problem constants (match setup_inputs) ----
static constexpr int ZD = 8, HDm = 93, WDm = 180, CD = 192;
static constexpr int NHEADS = 6, HDIM = 32;
static constexpr int NTOK = 144;              // 2*6*12
static constexpr int NWIN = 960;              // 4*16*15 windows
static constexpr int ZP = 8, HP = 96, WP = 180;
static constexpr int LTOK = ZD * HDm * WDm;   // 133920
static constexpr int LPAD = NWIN * NTOK;      // 138240

// ---------------- WMMA fragment helpers ----------------
// A-matrix 16x32 bf16: lane m(=lane&15), half=lane>>4:
//   elem e -> K = k0 + e + 16*half
static __device__ __forceinline__ v16bf load_fragA(const __bf16* base, int stride,
                                                   int row0, int k0, int lane) {
  int half = lane >> 4, m = lane & 15;
  const __bf16* p = base + (size_t)(row0 + m) * stride + k0 + half * 16;
  v8bf lo = *(const v8bf*)p;
  v8bf hi = *(const v8bf*)(p + 8);
  v16bf r;
#pragma unroll
  for (int i = 0; i < 8; i++) { r[i] = lo[i]; r[8 + i] = hi[i]; }
  return r;
}

static __device__ __forceinline__ v16bf load_fragA_f32(const float* base, int stride,
                                                       int row0, int k0, int lane) {
  int half = lane >> 4, m = lane & 15;
  const float* p = base + (size_t)(row0 + m) * stride + k0 + half * 16;
  v16bf r;
#pragma unroll
  for (int i = 0; i < 8; i++) { r[i] = (__bf16)p[i]; r[8 + i] = (__bf16)p[8 + i]; }
  return r;
}

// B-matrix 32x16 bf16: lane n(=lane&15), half=lane>>4: elem e -> K = e + 16*half.
// Fragment where B[k][n] = S[n0+n][k] (i.e. B = S^T, S row-major, row >= 32 elems):
static __device__ __forceinline__ v16bf load_fragB_rows(const __bf16* S, int stride,
                                                        int n0, int lane) {
  int half = lane >> 4, n = lane & 15;
  return *(const v16bf*)(S + (size_t)(n0 + n) * stride + half * 16);
}

// Pre-packed weight fragments: pack[tile][lane][e], tile = kt*CT+ct, 32B/lane.
static __device__ __forceinline__ v16bf load_fragB_pack(const __bf16* pack, int CT,
                                                        int kt, int ct, int lane) {
  return *(const v16bf*)(pack + ((size_t)(kt * CT + ct) * 32 + lane) * 16);
}

static __device__ __forceinline__ v8f wmma_bf16(v16bf a, v16bf b, v8f c) {
  return __builtin_amdgcn_wmma_f32_16x16x32_bf16(false, a, false, b, (short)0, c,
                                                 false, false);
}

// 16-lane reductions (stay within a half of the wave; xor masks 1..8 preserve bit4)
static __device__ __forceinline__ float red_max16(float v) {
#pragma unroll
  for (int m = 1; m < 16; m <<= 1) v = fmaxf(v, __shfl_xor(v, m, 32));
  return v;
}
static __device__ __forceinline__ float red_sum16(float v) {
#pragma unroll
  for (int m = 1; m < 16; m <<= 1) v += __shfl_xor(v, m, 32);
  return v;
}

// ---------------- K0: pack fp32 weights -> bf16 B-fragment layout ----------------
__global__ void pack_weights(const float* __restrict__ W, __bf16* __restrict__ dst,
                             int Ndim) {
  int tile = blockIdx.x;
  int CT = Ndim >> 4;
  int kt = tile / CT, ct = tile - kt * CT;
  int tid = threadIdx.x;          // 0..511
  int lane = tid >> 4, e = tid & 15;
  int half = lane >> 4, n = lane & 15;
  int k = kt * 32 + e + 16 * half;
  dst[(size_t)tile * 512 + lane * 16 + e] = (__bf16)W[(size_t)k * Ndim + ct * 16 + n];
}

// ---------------- K1: pad + roll + window-partition, fp32 -> bf16 ----------------
__global__ void gather_win(const float* __restrict__ x, __bf16* __restrict__ win) {
  int R = blockIdx.x;             // 0..LPAD-1 (window token row)
  int c = threadIdx.x;            // 0..191
  int w = R / NTOK, tok = R - w * NTOK;
  int zw = w / 240; int rem = w - zw * 240; int hw = rem / 15; int ww = rem - hw * 15;
  int iz = tok / 72; int r2 = tok - iz * 72; int ih = r2 / 12; int iw = r2 - ih * 12;
  int zp = zw * 2 + iz, hp = hw * 6 + ih, wp = ww * 12 + iw;   // rolled padded coords
  int zz = (zp + ZP - 1) % ZP;                                  // un-roll (shift 1,3,6)
  int hh = (hp + HP - 3) % HP;
  int w2 = (wp + WP - 6) % WP;
  float v = 0.f;
  if (hh >= 1 && hh <= 93) {                                    // un-pad H (p1=1,p2=2)
    int h = hh - 1;
    v = x[((size_t)(zz * HDm + h) * WDm + w2) * CD + c];
  }
  win[(size_t)R * CD + c] = (__bf16)v;
}

// ---- K2: QKV GEMM (138240x192 @ 192x576), 9 col-tiles per block for A reuse ----
__global__ __launch_bounds__(32) void qkv_gemm(const __bf16* __restrict__ win,
                                               const __bf16* __restrict__ wpack,
                                               const float* __restrict__ bias,
                                               __bf16* __restrict__ q,
                                               __bf16* __restrict__ k,
                                               __bf16* __restrict__ v) {
  int r0 = blockIdx.x * 16;
  int ctg = blockIdx.y * 9;       // column-tile group: {0..8, 9..17, 18..26, 27..35}
  int lane = threadIdx.x;
  __builtin_prefetch(wpack + (size_t)ctg * 512, 0, 0);
  v8f acc[9];
#pragma unroll
  for (int c9 = 0; c9 < 9; c9++) { v8f z{}; acc[c9] = z; }
#pragma unroll
  for (int kt = 0; kt < 6; kt++) {
    v16bf a = load_fragA(win, CD, r0, kt * 32, lane);
#pragma unroll
    for (int c9 = 0; c9 < 9; c9++) {
      v16bf b = load_fragB_pack(wpack, 36, kt, ctg + c9, lane);
      acc[c9] = wmma_bf16(a, b, acc[c9]);
    }
  }
  int half = lane >> 4, n = lane & 15;
  int w = r0 / NTOK; int tok0 = r0 - w * NTOK;    // 144 % 16 == 0 -> same window
#pragma unroll
  for (int c9 = 0; c9 < 9; c9++) {
    int ct = ctg + c9;
    int col = ct * 16 + n;
    float bcol = bias[col];
    int which = col / 192; int cc = col - which * 192;
    int head = cc >> 5; int d = cc & 31;
    __bf16* dst = (which == 0) ? q : ((which == 1) ? k : v);
    float scale = (which == 0) ? 0.17677669529663687f : 1.0f;  // 1/sqrt(32) into q
#pragma unroll
    for (int r = 0; r < 8; r++) {
      int tok = tok0 + r + 8 * half;
      float val = (acc[c9][r] + bcol) * scale;
      dst[((size_t)(w * NHEADS + head) * NTOK + tok) * HDIM + d] = (__bf16)val;
    }
  }
}

// ---- K3: fused windowed attention; one block per (head, window), all 9 row tiles ----
__global__ __launch_bounds__(32) void attn_kernel(const __bf16* __restrict__ q,
                                                  const __bf16* __restrict__ k,
                                                  const __bf16* __restrict__ v,
                                                  __bf16* __restrict__ out) {
  __shared__ __attribute__((aligned(32))) __bf16 vT[32 * 160];    // v transposed, K-pad
  __shared__ __attribute__((aligned(32))) __bf16 attnS[16 * 160]; // probs, K-pad
  __shared__ int region[NTOK];
  int head = blockIdx.x;
  int w = blockIdx.y;
  int lane = threadIdx.x;
  int half = lane >> 4, n = lane & 15;

  int zw = w / 240; int rem = w - zw * 240; int hw = rem / 15;
  for (int t = lane; t < NTOK; t += 32) {                 // shift-window mask regions
    int iz = t / 72; int r2 = t - iz * 72; int ih = r2 / 12;
    int zp = zw * 2 + iz, hp = hw * 6 + ih;
    int zc = (zp < 6) ? 0 : ((zp == 6) ? 1 : 2);
    int hc = (hp < 90) ? 0 : ((hp < 93) ? 1 : 2);
    region[t] = zc * 3 + hc;
  }
  const __bf16* vb = v + (size_t)(w * NHEADS + head) * NTOK * HDIM;
  for (int idx = lane; idx < NTOK * HDIM; idx += 32) {    // stage v transposed in LDS
    int tok = idx >> 5, d = idx & 31;
    vT[d * 160 + tok] = vb[idx];
  }
  for (int idx = lane; idx < 16 * 32; idx += 32) {        // zero K-pad rows 144..159
    int tok = 144 + (idx >> 5), d = idx & 31;
    vT[d * 160 + tok] = (__bf16)0.f;
  }
  for (int idx = lane; idx < 16 * 16; idx += 32) {        // zero attn pad cols
    int r = idx >> 4, cpos = 144 + (idx & 15);
    attnS[r * 160 + cpos] = (__bf16)0.f;
  }
  __syncthreads();

  const __bf16* qb = q + (size_t)(w * NHEADS + head) * NTOK * HDIM;
  const __bf16* kb = k + (size_t)(w * NHEADS + head) * NTOK * HDIM;
  // Preload all 9 k^T B-fragments once; reused by every row tile.
  v16bf kfr[9];
#pragma unroll
  for (int j = 0; j < 9; j++) kfr[j] = load_fragB_rows(kb, HDIM, j * 16, lane);

  for (int i = 0; i < 9; i++) {                           // 16-row score tiles
    v16bf aq = load_fragA(qb, HDIM, i * 16, 0, lane);     // hd=32 -> single K-step
    v8f acc[9];
#pragma unroll
    for (int j = 0; j < 9; j++) {
      v8f z{};
      acc[j] = wmma_bf16(aq, kfr[j], z);
    }
    // mask + softmax, row-wise (row M = r + 8*half lives in one 16-lane half)
#pragma unroll
    for (int r = 0; r < 8; r++) {
      int tr = i * 16 + r + 8 * half;
      int regr = region[tr];
      float mx = -3.0e38f;
#pragma unroll
      for (int j = 0; j < 9; j++) {
        int tc = j * 16 + n;
        float s = acc[j][r] + ((region[tc] == regr) ? 0.f : -10000.f);
        acc[j][r] = s;
        mx = fmaxf(mx, s);
      }
      mx = red_max16(mx);
      float sum = 0.f;
#pragma unroll
      for (int j = 0; j < 9; j++) {
        float p = __expf(acc[j][r] - mx);
        acc[j][r] = p; sum += p;
      }
      sum = red_sum16(sum);
      float inv = 1.f / sum;
#pragma unroll
      for (int j = 0; j < 9; j++)
        attnS[(r + 8 * half) * 160 + j * 16 + n] = (__bf16)(acc[j][r] * inv);
    }
    __syncthreads();
    // P @ V : K padded to 160, 5 K-steps, 2 column tiles of 16
#pragma unroll
    for (int dt = 0; dt < 2; dt++) {
      v8f o{};
#pragma unroll
      for (int kc = 0; kc < 5; kc++) {
        v16bf a = load_fragA(attnS, 160, 0, kc * 32, lane);
        v16bf b = *(const v16bf*)(vT + (dt * 16 + n) * 160 + kc * 32 + 16 * half);
        o = wmma_bf16(a, b, o);
      }
#pragma unroll
      for (int r = 0; r < 8; r++) {
        int tok = i * 16 + r + 8 * half;
        out[((size_t)w * NTOK + tok) * CD + head * HDIM + dt * 16 + n] = (__bf16)o[r];
      }
    }
    __syncthreads();   // attnS reused by next row tile
  }
}

// ------- K4: proj GEMM + reverse/unroll/unpad scatter + LayerNorm + residual -------
__global__ __launch_bounds__(32) void proj_kernel(const __bf16* __restrict__ ain,
                                                  const __bf16* __restrict__ wpack,
                                                  const float* __restrict__ pb,
                                                  const float* __restrict__ g1,
                                                  const float* __restrict__ b1,
                                                  const float* __restrict__ xin,
                                                  float* __restrict__ x1) {
  int r0 = blockIdx.x * 16;
  int lane = threadIdx.x;
  int half = lane >> 4, n = lane & 15;
  __builtin_prefetch(wpack, 0, 0);
  v16bf A[6];
#pragma unroll
  for (int kt = 0; kt < 6; kt++) A[kt] = load_fragA(ain, CD, r0, kt * 32, lane);
  v8f acc[12];
#pragma unroll
  for (int ct = 0; ct < 12; ct++) {
    v8f a{};
#pragma unroll
    for (int kt = 0; kt < 6; kt++) {
      v16bf b = load_fragB_pack(wpack, 12, kt, ct, lane);
      a = wmma_bf16(A[kt], b, a);
    }
    float bias = pb[ct * 16 + n];
#pragma unroll
    for (int r = 0; r < 8; r++) a[r] += bias;
    acc[ct] = a;
  }
  float g[12], bb[12];
#pragma unroll
  for (int ct = 0; ct < 12; ct++) { g[ct] = g1[ct * 16 + n]; bb[ct] = b1[ct * 16 + n]; }
  int w = r0 / NTOK; int tok0 = r0 - w * NTOK;
  int zw = w / 240; int rem = w - zw * 240; int hw = rem / 15; int ww = rem - hw * 15;
#pragma unroll
  for (int r = 0; r < 8; r++) {
    float s = 0.f;
#pragma unroll
    for (int ct = 0; ct < 12; ct++) s += acc[ct][r];
    s = red_sum16(s);
    float mean = s * (1.f / 192.f);
    float var = 0.f;
#pragma unroll
    for (int ct = 0; ct < 12; ct++) { float d = acc[ct][r] - mean; var += d * d; }
    var = red_sum16(var);
    float rstd = rsqrtf(var * (1.f / 192.f) + 1e-5f);
    int tok = tok0 + r + 8 * half;
    int iz = tok / 72; int rr = tok - iz * 72; int ih = rr / 12; int iw = rr - ih * 12;
    int zp = zw * 2 + iz, hp = hw * 6 + ih, wp2 = ww * 12 + iw;
    int zz = (zp + ZP - 1) % ZP;
    int hh = (hp + HP - 3) % HP;
    int w2 = (wp2 + WP - 6) % WP;
    if (hh >= 1 && hh <= 93) {
      int orow = (zz * HDm + (hh - 1)) * WDm + w2;
#pragma unroll
      for (int ct = 0; ct < 12; ct++) {
        int c = ct * 16 + n;
        float val = (acc[ct][r] - mean) * rstd * g[ct] + bb[ct];
        x1[(size_t)orow * CD + c] = xin[(size_t)orow * CD + c] + val;
      }
    }
  }
}

// ------- K5: fused MLP (x1 @ W1 -> gelu in LDS -> @ W2) + LayerNorm + residual ------
__global__ __launch_bounds__(32) void mlp_kernel(const float* __restrict__ x1,
                                                 const __bf16* __restrict__ w1p,
                                                 const float* __restrict__ b1v,
                                                 const __bf16* __restrict__ w2p,
                                                 const float* __restrict__ b2v,
                                                 const float* __restrict__ g2,
                                                 const float* __restrict__ bb2,
                                                 float* __restrict__ outp) {
  __shared__ __attribute__((aligned(32))) __bf16 hbuf[16 * 768];
  int r0 = blockIdx.x * 16;
  int lane = threadIdx.x;
  int half = lane >> 4, n = lane & 15;
  __builtin_prefetch(w1p, 0, 0);
  __builtin_prefetch(w2p, 0, 0);
  v16bf A1[6];
#pragma unroll
  for (int kt = 0; kt < 6; kt++) A1[kt] = load_fragA_f32(x1, CD, r0, kt * 32, lane);
  for (int ct = 0; ct < 48; ct++) {
    v8f a{};
#pragma unroll
    for (int kt = 0; kt < 6; kt++) {
      v16bf b = load_fragB_pack(w1p, 48, kt, ct, lane);
      a = wmma_bf16(A1[kt], b, a);
    }
    float bias = b1v[ct * 16 + n];
#pragma unroll
    for (int r = 0; r < 8; r++) {
      float xv = a[r] + bias;
      float ge = 0.5f * xv * (1.f + erff(xv * 0.70710678118654752f));  // exact gelu
      hbuf[(r + 8 * half) * 768 + ct * 16 + n] = (__bf16)ge;
    }
  }
  __syncthreads();
  v8f acc[12];
#pragma unroll
  for (int ct = 0; ct < 12; ct++) { v8f z{}; acc[ct] = z; }
  for (int kt = 0; kt < 24; kt++) {
    v16bf a2 = load_fragA(hbuf, 768, 0, kt * 32, lane);
#pragma unroll
    for (int ct = 0; ct < 12; ct++) {
      v16bf b = load_fragB_pack(w2p, 12, kt, ct, lane);
      acc[ct] = wmma_bf16(a2, b, acc[ct]);
    }
  }
  float g[12], bbv[12], bias2[12];
#pragma unroll
  for (int ct = 0; ct < 12; ct++) {
    g[ct] = g2[ct * 16 + n]; bbv[ct] = bb2[ct * 16 + n]; bias2[ct] = b2v[ct * 16 + n];
  }
#pragma unroll
  for (int r = 0; r < 8; r++) {
    float y[12]; float s = 0.f;
#pragma unroll
    for (int ct = 0; ct < 12; ct++) { y[ct] = acc[ct][r] + bias2[ct]; s += y[ct]; }
    s = red_sum16(s);
    float mean = s * (1.f / 192.f);
    float var = 0.f;
#pragma unroll
    for (int ct = 0; ct < 12; ct++) { float d = y[ct] - mean; var += d * d; }
    var = red_sum16(var);
    float rstd = rsqrtf(var * (1.f / 192.f) + 1e-5f);
    int row = r0 + r + 8 * half;
#pragma unroll
    for (int ct = 0; ct < 12; ct++) {
      int c = ct * 16 + n;
      float ln = (y[ct] - mean) * rstd * g[ct] + bbv[ct];
      outp[(size_t)row * CD + c] = x1[(size_t)row * CD + c] + ln;
    }
  }
}

// ---------------- host launch ----------------
extern "C" void kernel_launch(void* const* d_in, const int* in_sizes, int n_in,
                              void* d_out, int out_size, void* d_ws, size_t ws_size,
                              hipStream_t stream) {
  const float* x     = (const float*)d_in[0];
  const float* n1g   = (const float*)d_in[1];
  const float* n1b   = (const float*)d_in[2];
  const float* n2g   = (const float*)d_in[3];
  const float* n2b   = (const float*)d_in[4];
  const float* qkvw  = (const float*)d_in[5];
  const float* qkvb  = (const float*)d_in[6];
  const float* projw = (const float*)d_in[7];
  const float* projb = (const float*)d_in[8];
  const float* w1    = (const float*)d_in[9];
  const float* b1    = (const float*)d_in[10];
  const float* w2    = (const float*)d_in[11];
  const float* b2    = (const float*)d_in[12];

  char* ws = (char*)d_ws;
  size_t off = 0;
  auto alloc = [&](size_t bytes) -> char* {
    char* p = ws + off;
    off = (off + bytes + 255) & ~(size_t)255;
    return p;
  };
  __bf16* win = (__bf16*)alloc((size_t)LPAD * CD * 2);   // 53 MB (reused as attn out)
  __bf16* qb  = (__bf16*)alloc((size_t)LPAD * CD * 2);   // 53 MB
  __bf16* kb  = (__bf16*)alloc((size_t)LPAD * CD * 2);   // 53 MB
  __bf16* vb  = (__bf16*)alloc((size_t)LPAD * CD * 2);   // 53 MB
  float*  x1  = (float*)alloc((size_t)LTOK * CD * 4);    // 103 MB
  __bf16* pq  = (__bf16*)alloc((size_t)6 * 36 * 512 * 2);
  __bf16* pp  = (__bf16*)alloc((size_t)6 * 12 * 512 * 2);
  __bf16* p1  = (__bf16*)alloc((size_t)6 * 48 * 512 * 2);
  __bf16* p2  = (__bf16*)alloc((size_t)24 * 12 * 512 * 2);
  __bf16* attn_out = win;  // win is dead after qkv_gemm; alias as attention output

  pack_weights<<<6 * 36, 512, 0, stream>>>(qkvw, pq, 576);
  pack_weights<<<6 * 12, 512, 0, stream>>>(projw, pp, 192);
  pack_weights<<<6 * 48, 512, 0, stream>>>(w1, p1, 768);
  pack_weights<<<24 * 12, 512, 0, stream>>>(w2, p2, 192);

  gather_win<<<LPAD, CD, 0, stream>>>(x, win);

  dim3 gq(LPAD / 16, 4);                 // 9 column tiles per block
  qkv_gemm<<<gq, 32, 0, stream>>>(win, pq, qkvb, qb, kb, vb);

  dim3 ga(NHEADS, NWIN);                 // all 9 row tiles inside one block
  attn_kernel<<<ga, 32, 0, stream>>>(qb, kb, vb, attn_out);

  proj_kernel<<<LPAD / 16, 32, 0, stream>>>(attn_out, pp, projb, n1g, n1b, x, x1);

  mlp_kernel<<<LTOK / 16, 32, 0, stream>>>(x1, p1, b1, p2, b2, n2g, n2b,
                                           (float*)d_out);
}
